// FusedBlockTorch_51135880626819
// MI455X (gfx1250) — matrix operand order repaired and checked by reading, not compiled, not run on hardware
//
#include <hip/hip_runtime.h>
#include <hip/hip_bf16.h>

#define HID 2048
#define TOK 16384

typedef __attribute__((ext_vector_type(16))) __bf16  v16bf;
typedef __attribute__((ext_vector_type(8)))  float   v8f;
typedef unsigned int u32x4 __attribute__((ext_vector_type(4)));

union BFrag {
    u32x4 q[2];
    v16bf v;
};

// round-to-nearest-even fp32 -> bf16 (bit pattern)
__device__ __forceinline__ unsigned short f2bf(float f) {
    unsigned int u = __float_as_uint(f);
    unsigned int r = u + 0x7FFFu + ((u >> 16) & 1u);
    return (unsigned short)(r >> 16);
}

// ---------------------------------------------------------------------------
// Kernel 1: per-token LayerNorm, emit bf16 normalized activations.
// One 256-thread block per token; each thread handles 8 elements.
// ---------------------------------------------------------------------------
__global__ __launch_bounds__(256) void ln_bf16_kernel(
    const float* __restrict__ x,
    const float* __restrict__ gamma,
    const float* __restrict__ beta,
    unsigned short* __restrict__ xn)
{
    const int t   = blockIdx.x;
    const int tid = threadIdx.x;
    const int i0  = tid * 8;
    const float* row = x + (size_t)t * HID;

    const float4* p4 = reinterpret_cast<const float4*>(row + i0);
    float4 f0 = p4[0];
    float4 f1 = p4[1];

    float s  = f0.x + f0.y + f0.z + f0.w + f1.x + f1.y + f1.z + f1.w;
    float sq = f0.x*f0.x + f0.y*f0.y + f0.z*f0.z + f0.w*f0.w +
               f1.x*f1.x + f1.y*f1.y + f1.z*f1.z + f1.w*f1.w;

    __shared__ float red_s[256];
    __shared__ float red_q[256];
    red_s[tid] = s;
    red_q[tid] = sq;
    __syncthreads();
    #pragma unroll
    for (int st = 128; st > 0; st >>= 1) {
        if (tid < st) {
            red_s[tid] += red_s[tid + st];
            red_q[tid] += red_q[tid + st];
        }
        __syncthreads();
    }
    const float inv_h = 1.0f / (float)HID;
    const float mu    = red_s[0] * inv_h;
    const float var   = red_q[0] * inv_h - mu * mu;
    const float rstd  = rsqrtf(var + 1e-5f);

    const float4* g4 = reinterpret_cast<const float4*>(gamma + i0);
    const float4* b4 = reinterpret_cast<const float4*>(beta + i0);
    float4 g0 = g4[0], g1 = g4[1];
    float4 b0 = b4[0], b1 = b4[1];

    float v[8];
    v[0] = (f0.x - mu) * rstd * g0.x + b0.x;
    v[1] = (f0.y - mu) * rstd * g0.y + b0.y;
    v[2] = (f0.z - mu) * rstd * g0.z + b0.z;
    v[3] = (f0.w - mu) * rstd * g0.w + b0.w;
    v[4] = (f1.x - mu) * rstd * g1.x + b1.x;
    v[5] = (f1.y - mu) * rstd * g1.y + b1.y;
    v[6] = (f1.z - mu) * rstd * g1.z + b1.z;
    v[7] = (f1.w - mu) * rstd * g1.w + b1.w;

    u32x4 pk;
    pk.x = (unsigned)f2bf(v[0]) | ((unsigned)f2bf(v[1]) << 16);
    pk.y = (unsigned)f2bf(v[2]) | ((unsigned)f2bf(v[3]) << 16);
    pk.z = (unsigned)f2bf(v[4]) | ((unsigned)f2bf(v[5]) << 16);
    pk.w = (unsigned)f2bf(v[6]) | ((unsigned)f2bf(v[7]) << 16);
    *reinterpret_cast<u32x4*>(xn + (size_t)t * HID + i0) = pk;
}

// ---------------------------------------------------------------------------
// Kernel 2: W fp32 [2048][2048] row-major -> bf16 row-major.
// ---------------------------------------------------------------------------
__global__ __launch_bounds__(256) void cvt_w_kernel(
    const float* __restrict__ W,
    unsigned short* __restrict__ Wb)
{
    const size_t i0 = ((size_t)blockIdx.x * 256 + threadIdx.x) * 8;
    const float4* p4 = reinterpret_cast<const float4*>(W + i0);
    float4 f0 = p4[0];
    float4 f1 = p4[1];
    u32x4 pk;
    pk.x = (unsigned)f2bf(f0.x) | ((unsigned)f2bf(f0.y) << 16);
    pk.y = (unsigned)f2bf(f0.z) | ((unsigned)f2bf(f0.w) << 16);
    pk.z = (unsigned)f2bf(f1.x) | ((unsigned)f2bf(f1.y) << 16);
    pk.w = (unsigned)f2bf(f1.z) | ((unsigned)f2bf(f1.w) << 16);
    *reinterpret_cast<u32x4*>(Wb + i0) = pk;
}

// ---------------------------------------------------------------------------
// Kernel 3: WMMA bf16 GEMM  y = xn @ W^T + b, then exact-erf GELU.
// Block = 256 threads = 8 wave32s. Block tile 128(M) x 128(N).
// Wave tile: 32(M) x 64(N) -> 2 M-tiles x 4 N-tiles = 8 accumulators.
// K pipelined with register ping-pong; pointers advance by 64 elements per
// unrolled iteration so every load uses an immediate offset.
//
// Fragment layouts (CDNA5 ISA 7.12.2, 16-bit, wave32):
//  A 16x32: lanes 0-15 -> row M=lane,   K 0..7 (q0) and 16..23 (q1)
//           lanes 16-31-> row M=lane-16,K 8..15(q0) and 24..31 (q1)
//  B 32x16: lanes 0-15 -> col N=lane,   K 0..7 (q0) and 16..23 (q1)
//           lanes 16-31-> col N=lane-16,K 8..15(q0) and 24..31 (q1)
//  C/D 16x16 f32: VGPR r, lanes 0-15 -> M=r, lanes 16-31 -> M=r+8; N=lane&15
// ---------------------------------------------------------------------------
struct Frags {
    BFrag a[2];
    BFrag b[4];
};

// koff is a compile-time element offset inside the unrolled loop body.
__device__ __forceinline__ void load_step(Frags& f,
                                          const unsigned short* __restrict__ aP0,
                                          const unsigned short* __restrict__ aP1,
                                          const unsigned short* __restrict__ bP,
                                          int koff)
{
    f.a[0].q[0] = *reinterpret_cast<const u32x4*>(aP0 + koff);
    f.a[0].q[1] = *reinterpret_cast<const u32x4*>(aP0 + koff + 16);
    f.a[1].q[0] = *reinterpret_cast<const u32x4*>(aP1 + koff);
    f.a[1].q[1] = *reinterpret_cast<const u32x4*>(aP1 + koff + 16);
    #pragma unroll
    for (int nt = 0; nt < 4; ++nt) {
        const unsigned short* bp = bP + nt * 16 * HID + koff;
        f.b[nt].q[0] = *reinterpret_cast<const u32x4*>(bp);
        f.b[nt].q[1] = *reinterpret_cast<const u32x4*>(bp + 16);
    }
}

__device__ __forceinline__ void mma_step(v8f acc[2][4], const Frags& f)
{
    #pragma unroll
    for (int mt = 0; mt < 2; ++mt) {
        #pragma unroll
        for (int nt = 0; nt < 4; ++nt) {
            acc[mt][nt] = __builtin_amdgcn_wmma_f32_16x16x32_bf16(
                /*neg_a=*/false, f.a[mt].v, /*neg_b=*/false, f.b[nt].v,
                /*c_mod=*/(short)0, acc[mt][nt],
                /*reuse_a=*/false, /*reuse_b=*/false);
        }
    }
}

__global__ __launch_bounds__(256, 1) void gemm_wmma_gelu_kernel(
    const unsigned short* __restrict__ Xn,
    const unsigned short* __restrict__ Wb,
    const float* __restrict__ bias,
    float* __restrict__ out)
{
    const int lane   = threadIdx.x & 31;
    const int wave   = threadIdx.x >> 5;
    const int lane16 = lane & 15;
    const int laneHi = lane >> 4;        // 0 or 1

    const int mWave = wave & 3;          // 4 M-waves
    const int nWave = wave >> 2;         // 2 N-waves
    const int mBase = blockIdx.y * 128 + mWave * 32;   // wave covers 32 rows
    const int nBase = blockIdx.x * 128 + nWave * 64;   // wave covers 64 cols

    // Per-lane A bases for the wave's two 16-row M-tiles.
    const unsigned short* aP0 = Xn + (size_t)(mBase +      lane16) * HID + laneHi * 8;
    const unsigned short* aP1 = Xn + (size_t)(mBase + 16 + lane16) * HID + laneHi * 8;
    // Per-lane B base: W row (nBase+lane16) (= output col), K offset laneHi*8.
    const unsigned short* bP  = Wb + (size_t)(nBase + lane16) * HID + laneHi * 8;

    v8f acc[2][4] = {};

    constexpr int NSTEP = HID / 32;      // 64 K-steps

    Frags f0, f1;
    load_step(f0, aP0, aP1, bP, 0);

    // Ping-pong pipeline: loads for step s+1 issue before WMMAs of step s.
    // Pointers advance 64 elements (2 K-steps) per iteration; all load
    // offsets inside the body are immediates.
    #pragma clang loop unroll(disable)
    for (int it = 0; it < (NSTEP - 2) / 2; ++it) {
        load_step(f1, aP0, aP1, bP, 32);     // step s+1
        mma_step(acc, f0);
        load_step(f0, aP0, aP1, bP, 64);     // step s+2
        mma_step(acc, f1);
        aP0 += 64; aP1 += 64; bP += 64;
    }
    // Tail: f0 holds step NSTEP-2; load and run final step NSTEP-1.
    load_step(f1, aP0, aP1, bP, 32);
    mma_step(acc, f0);
    mma_step(acc, f1);

    // Epilogue: bias + exact GELU (0.5*y*(1+erf(y/sqrt(2)))), fp32 out.
    const float kInvSqrt2 = 0.70710678118654752440f;
    #pragma unroll
    for (int mt = 0; mt < 2; ++mt) {
        #pragma unroll
        for (int nt = 0; nt < 4; ++nt) {
            const int col = nBase + nt * 16 + lane16;
            const float bb = bias[col];
            #pragma unroll
            for (int r = 0; r < 8; ++r) {
                const int row = mBase + mt * 16 + r + laneHi * 8;
                float y = acc[mt][nt][r] + bb;
                float g = 0.5f * y * (1.0f + erff(y * kInvSqrt2));
                __builtin_nontemporal_store(g, out + (size_t)row * HID + col);
            }
        }
    }
}

// ---------------------------------------------------------------------------
// Launch
// ---------------------------------------------------------------------------
extern "C" void kernel_launch(void* const* d_in, const int* in_sizes, int n_in,
                              void* d_out, int out_size, void* d_ws, size_t ws_size,
                              hipStream_t stream) {
    const float* x     = (const float*)d_in[0];
    const float* gamma = (const float*)d_in[1];
    const float* beta  = (const float*)d_in[2];
    const float* W     = (const float*)d_in[3];
    const float* b     = (const float*)d_in[4];
    float* out = (float*)d_out;

    unsigned short* xn = (unsigned short*)d_ws;                  // 16384*2048 bf16 = 64 MB
    unsigned short* wb = xn + (size_t)TOK * HID;                 // 2048*2048  bf16 =  8 MB

    ln_bf16_kernel<<<TOK, 256, 0, stream>>>(x, gamma, beta, xn);
    cvt_w_kernel<<<(HID * HID) / (256 * 8), 256, 0, stream>>>(W, wb);

    dim3 grid(HID / 128, TOK / 128);   // (16, 128)
    gemm_wmma_gelu_kernel<<<grid, 256, 0, stream>>>(xn, wb, b, out);
}
